// LookupFreeQuantization_55860344652140
// MI455X (gfx1250) — matrix operand order repaired and checked by reading, not compiled, not run on hardware
//
#include <hip/hip_runtime.h>
#include <hip/hip_bf16.h>

// LFQ forward for MI455X (gfx1250, wave32).
// Bandwidth-bound: ~0.9 GB traffic -> ~40us at 23.3 TB/s peak.
// WMMA f16 16x16x32 for the down-projection (K=1024), WMMA f32 16x16x4 for
// the up-projection (K=12). token_log_probs via exact LUT factorization:
//   tlp[t,c] = sum_i bit_i(c)*4*z_i*2^i - sum_i softplus(4*z_i*2^i).
// Streaming outputs use non-temporal stores to keep the 192MB L2 for reuse.

#define LATENT 1024
#define DBITS  12
#define TOKENS 32768   // B*S = 8*4096

typedef __attribute__((ext_vector_type(16))) _Float16 v16h;
typedef __attribute__((ext_vector_type(8)))  _Float16 v8h;
typedef __attribute__((ext_vector_type(8)))  float    v8f;
typedef __attribute__((ext_vector_type(4)))  float    f32x4;
typedef __attribute__((ext_vector_type(2)))  float    v2f;

#if defined(__has_builtin)
#if __has_builtin(__builtin_amdgcn_wmma_f32_16x16x4_f32)
#define HAVE_WMMA_F32X4 1
#endif
#endif
#ifndef HAVE_WMMA_F32X4
#define HAVE_WMMA_F32X4 0
#endif

// Branchless tanh: 1 - 2/(1+exp(2p)).  exp->+inf => 1, exp->0 => -1.
// One v_exp_f32 + one v_rcp_f32; avoids libm's exec-masked expansion.
__device__ __forceinline__ float fast_tanh(float p) {
    const float e = __expf(2.0f * p);
    return 1.0f - 2.0f * __builtin_amdgcn_rcpf(e + 1.0f);
}

// ---------------------------------------------------------------------------
// Kernel 1: z = tanh(x @ Wd + bd); also tokens, commit-loss partials, z->ws.
// 256 threads = 8 waves; wave handles 16 tokens; block handles 128 tokens.
// ---------------------------------------------------------------------------
__global__ __launch_bounds__(256) void lfq_down_kernel(
    const float* __restrict__ x, const float* __restrict__ Wd,
    const float* __restrict__ bd, float* __restrict__ zws,
    float* __restrict__ tok_out, float* __restrict__ acc)
{
    // Wd transposed in LDS as f16: [16 cols][1024 rows], row-padded to 1032
    // halfs so lane strides hit distinct bank groups (1032*2B = 516 dwords).
    __shared__ _Float16 ldsWdT[16 * 1032];
    __shared__ float    ldsbd[16];
    const int tid = threadIdx.x;
    for (int i = tid; i < 16 * LATENT; i += 256) {
        const int n = i >> 10, k = i & (LATENT - 1);
        ldsWdT[n * 1032 + k] = (_Float16)((n < DBITS) ? Wd[k * DBITS + n] : 0.0f);
    }
    if (tid < 16) ldsbd[tid] = (tid < DBITS) ? bd[tid] : 0.0f;
    __syncthreads();

    const int wave = tid >> 5;
    const int lane = tid & 31;
    const int half = lane >> 4;   // 0: lanes 0-15, 1: lanes 16-31
    const int l16  = lane & 15;
    const int tok0 = blockIdx.x * 128 + wave * 16;

    // A-matrix row for this lane (16-bit A 16x32: both lane halves cover M=0..15)
    const float* __restrict__ xrow = x + (size_t)(tok0 + l16) * LATENT;
    const _Float16* __restrict__ bcol = ldsWdT + l16 * 1032;

    v8f c = {};
    for (int kb = 0; kb < LATENT; kb += 32) {
        if (kb + 288 <= LATENT)
            __builtin_prefetch(xrow + kb + 256, 0, 0);   // global_prefetch_b8

        // A frag: lane half h holds K = {kb+8h..kb+8h+7, kb+16+8h..kb+16+8h+7}
        const int ka = kb + half * 8;
        const f32x4 p0 = __builtin_nontemporal_load((const f32x4*)(xrow + ka));
        const f32x4 p1 = __builtin_nontemporal_load((const f32x4*)(xrow + ka + 4));
        const f32x4 q0 = __builtin_nontemporal_load((const f32x4*)(xrow + ka + 16));
        const f32x4 q1 = __builtin_nontemporal_load((const f32x4*)(xrow + ka + 20));
        v16h a;
        #pragma unroll
        for (int j = 0; j < 4; ++j) {
            a[j]      = (_Float16)p0[j];
            a[4 + j]  = (_Float16)p1[j];
            a[8 + j]  = (_Float16)q0[j];
            a[12 + j] = (_Float16)q1[j];
        }

        // B frag: lane n = l16 reads K = {kb+8h..+7, kb+16+8h..+7} (two ds_load_b128)
        const int kbase = kb + half * 8;
        const v8h blo = *(const v8h*)(bcol + kbase);
        const v8h bhi = *(const v8h*)(bcol + kbase + 16);
        v16h b;
        #pragma unroll
        for (int j = 0; j < 8; ++j) { b[j] = blo[j]; b[8 + j] = bhi[j]; }

        c = __builtin_amdgcn_wmma_f32_16x16x32_f16(
                false, a, false, b, (short)0, c, false, false);
    }

    // C/D layout: lane l16 = N, VGPR r -> M = r + 8*half
    float partial = 0.0f;
    const int n = l16;
    const float bias = ldsbd[n];
    #pragma unroll
    for (int r = 0; r < 8; ++r) {
        const int m = r + half * 8;
        float z = fast_tanh(c[r] + bias);
        if (n >= DBITS) z = 0.0f;                     // keep padding exactly zero
        zws[(size_t)(tok0 + m) * 16 + n] = z;         // cacheable: re-read by K2/K3
        if (n < DBITS) {
            const float t = 1.0f - fabsf(z);          // (z - sign(z))^2
            partial += t * t;
        }
        // Pack token bits per row via wave32 ballot (bit i lives on lane n=i)
        const unsigned mask = (unsigned)__ballot(z > 0.0f);
        if (lane == 0)  tok_out[tok0 + r]     = (float)(mask & 0xFFFu);
        if (lane == 16) tok_out[tok0 + 8 + r] = (float)((mask >> 16) & 0xFFFu);
    }

    // Wave-reduce commit-loss partial, one atomic per wave.
    for (int off = 16; off > 0; off >>= 1)
        partial += __shfl_down(partial, off, 32);
    if (lane == 0) atomicAdd(acc, partial);
}

// ---------------------------------------------------------------------------
// Kernel 2: z_up = z @ Wu + bu ; z_q_up = sign(z) @ Wu + bu.
// Primary: f32 WMMA 16x16x4 (K=12 in 3 steps). Fallback: f16 WMMA 16x16x32.
// Block = 8 waves; wave covers 16 tokens x 64 N; 2 blocks per token group.
// ---------------------------------------------------------------------------
__global__ __launch_bounds__(256) void lfq_up_kernel(
    const float* __restrict__ zws, const float* __restrict__ Wu,
    const float* __restrict__ bu, float* __restrict__ zup,
    float* __restrict__ zqup)
{
    __shared__ float ldsWu[DBITS * LATENT];  // 48 KB
    __shared__ float ldsbu[LATENT];          // 4 KB
    const int tid = threadIdx.x;
    for (int i = tid; i < DBITS * LATENT; i += 256) ldsWu[i] = Wu[i];
    for (int i = tid; i < LATENT; i += 256)        ldsbu[i] = bu[i];
    __syncthreads();

    const int wave = tid >> 5;
    const int lane = tid & 31;
    const int half = lane >> 4;
    const int l16  = lane & 15;
    const int g    = blockIdx.x >> 1;
    const int nh   = blockIdx.x & 1;
    const int tok0 = g * 16;
    const int n0   = nh * 512 + wave * 64;

    const float* __restrict__ zr = zws + (size_t)(tok0 + l16) * 16;

#if HAVE_WMMA_F32X4
    v8f cz[4]; v8f cq[4];
    #pragma unroll
    for (int t = 0; t < 4; ++t) { cz[t] = (v8f){}; cq[t] = (v8f){}; }

    #pragma unroll
    for (int ks = 0; ks < 3; ++ks) {          // K = 12 = 3 x 4 (no padding)
        // f32 A 16x4: lanes 0-15 K={4ks,4ks+1}, lanes 16-31 K={4ks+2,4ks+3}
        const int k0 = ks * 4 + half * 2;
        const v2f zz = *(const v2f*)(zr + k0);
        v2f a  = { zz[0], zz[1] };
        v2f aq = { (zz[0] > 0.0f) ? 1.0f : -1.0f, (zz[1] > 0.0f) ? 1.0f : -1.0f };
        #pragma unroll
        for (int t = 0; t < 4; ++t) {
            const int nn = n0 + t * 16 + l16;
            v2f b = { ldsWu[k0 * LATENT + nn], ldsWu[(k0 + 1) * LATENT + nn] };
            cz[t] = __builtin_amdgcn_wmma_f32_16x16x4_f32(
                        false, a,  false, b, (short)0, cz[t], false, false);
            cq[t] = __builtin_amdgcn_wmma_f32_16x16x4_f32(
                        false, aq, false, b, (short)0, cq[t], false, false);
        }
    }
    #pragma unroll
    for (int t = 0; t < 4; ++t) {
        const int nn = n0 + t * 16 + l16;
        const float bv = ldsbu[nn];
        #pragma unroll
        for (int r = 0; r < 8; ++r) {
            const int m = r + half * 8;
            const size_t off = (size_t)(tok0 + m) * LATENT + nn;
            __builtin_nontemporal_store(cz[t][r] + bv, &zup[off]);
            __builtin_nontemporal_store(cq[t][r] + bv, &zqup[off]);
        }
    }
#else
    // f16 fallback: single 16x16x32 WMMA per n-tile, K padded with zeros.
    v16h a = {}, aq = {};
    #pragma unroll
    for (int j = 0; j < 8; ++j) {
        const int k = half * 8 + j;            // K in 0..15; K>=16 stays zero
        const float zv = (k < DBITS) ? zr[k] : 0.0f;
        a[j]  = (_Float16)zv;
        aq[j] = (_Float16)((k < DBITS) ? ((zv > 0.0f) ? 1.0f : -1.0f) : 0.0f);
    }
    #pragma unroll
    for (int t = 0; t < 4; ++t) {
        const int nn = n0 + t * 16 + l16;
        v16h b;
        #pragma unroll
        for (int j = 0; j < 16; ++j) {
            const int k = half * 16 + j;
            b[j] = (_Float16)((k < DBITS) ? ldsWu[k * LATENT + nn] : 0.0f);
        }
        v8f cz = {}, cq = {};
        cz = __builtin_amdgcn_wmma_f32_16x16x32_f16(false, a,  false, b, (short)0, cz, false, false);
        cq = __builtin_amdgcn_wmma_f32_16x16x32_f16(false, aq, false, b, (short)0, cq, false, false);
        const float bv = ldsbu[nn];
        #pragma unroll
        for (int r = 0; r < 8; ++r) {
            const int m = r + half * 8;
            const size_t off = (size_t)(tok0 + m) * LATENT + nn;
            __builtin_nontemporal_store(cz[r] + bv, &zup[off]);
            __builtin_nontemporal_store(cq[r] + bv, &zqup[off]);
        }
    }
#endif
}

// ---------------------------------------------------------------------------
// Kernel 3: token_log_probs via exact LUT factorization. One wave per token,
// LUTs (2 x 64 floats) in LDS, output as wide NT b128 stores (512 MB total).
// ---------------------------------------------------------------------------
__global__ __launch_bounds__(256) void lfq_logprob_kernel(
    const float* __restrict__ zws, float* __restrict__ tlp)
{
    __shared__ float L[8][128];   // per wave: [0..63]=low-6-bit LUT, [64..127]=high
    const int tid  = threadIdx.x;
    const int wave = tid >> 5;
    const int lane = tid & 31;
    const int t    = blockIdx.x * 8 + wave;

    const float* __restrict__ zrow = zws + (size_t)t * 16;
    float d[DBITS];
    float lse = 0.0f;
    #pragma unroll
    for (int i = 0; i < DBITS; ++i) {
        const float di = 4.0f * zrow[i] * (float)(1 << i);
        d[i] = di;
        // stable softplus: max(x,0) + log1p(exp(-|x|))
        lse += fmaxf(di, 0.0f) + log1pf(__expf(-fabsf(di)));
    }
    #pragma unroll
    for (int rep = 0; rep < 2; ++rep) {
        const int j = lane + rep * 32;
        float s0 = 0.0f, s1 = 0.0f;
        #pragma unroll
        for (int i = 0; i < 6; ++i) {
            if (j & (1 << i)) { s0 += d[i]; s1 += d[i + 6]; }
        }
        L[wave][j]      = s0;
        L[wave][64 + j] = s1;
    }
    __syncthreads();

    f32x4* __restrict__ out = (f32x4*)(tlp + (size_t)t * 4096);
    const f32x4* __restrict__ L0v = (const f32x4*)&L[wave][0];
    #pragma unroll 4
    for (int it = 0; it < 32; ++it) {
        const int cbase = it * 128 + lane * 4;       // 4 codes, same high-6 block
        const float hi  = L[wave][64 + (cbase >> 6)] - lse;
        f32x4 lo = L0v[(cbase & 63) >> 2];
        f32x4 o  = { lo[0] + hi, lo[1] + hi, lo[2] + hi, lo[3] + hi };
        __builtin_nontemporal_store(o, out + (cbase >> 2));
    }
}

__global__ void lfq_zero_acc(float* acc) {
    if (threadIdx.x == 0 && blockIdx.x == 0) acc[0] = 0.0f;
}

__global__ void lfq_finalize(const float* __restrict__ acc, float* __restrict__ loss) {
    if (threadIdx.x == 0 && blockIdx.x == 0)
        loss[0] = acc[0] / (float)((size_t)TOKENS * DBITS);
}

extern "C" void kernel_launch(void* const* d_in, const int* in_sizes, int n_in,
                              void* d_out, int out_size, void* d_ws, size_t ws_size,
                              hipStream_t stream) {
    (void)in_sizes; (void)n_in; (void)out_size; (void)ws_size;
    const float* x  = (const float*)d_in[0];
    const float* Wd = (const float*)d_in[1];
    const float* bd = (const float*)d_in[2];
    const float* Wu = (const float*)d_in[3];
    const float* bu = (const float*)d_in[4];

    float* acc = (float*)d_ws;                       // commit-loss accumulator
    float* zws = (float*)((char*)d_ws + 256);        // z padded [TOKENS][16] f32 (2 MB)

    float* out   = (float*)d_out;
    float* tok   = out;                                              // 32768
    float* zup   = tok  + (size_t)TOKENS;                            // 33554432
    float* zqup  = zup  + (size_t)TOKENS * LATENT;                   // 33554432
    float* tlp   = zqup + (size_t)TOKENS * LATENT;                   // 134217728
    float* loss  = tlp  + (size_t)TOKENS * 4096;                     // 1

    lfq_zero_acc<<<1, 64, 0, stream>>>(acc);
    lfq_down_kernel<<<TOKENS / 128, 256, 0, stream>>>(x, Wd, bd, zws, tok, acc);
    lfq_finalize<<<1, 64, 0, stream>>>(acc, loss);
    lfq_up_kernel<<<(TOKENS / 16) * 2, 256, 0, stream>>>(zws, Wu, bu, zup, zqup);
    lfq_logprob_kernel<<<TOKENS / 8, 256, 0, stream>>>(zws, tlp);
}
